// LoggedSelfAttention_15659450761531
// MI455X (gfx1250) — matrix operand order repaired and checked by reading, not compiled, not run on hardware
//
#include <hip/hip_runtime.h>

// ---------------------------------------------------------------------------
// Self-attention (B=2, N=4096, D=512, h=8, d=64) for gfx1250 (MI455X).
// bf16 WMMA + fp32 accumulation, flash-attention streaming softmax,
// TDM (tensor_load_to_lds) double-buffered K/V staging, pre-converted bf16
// operands so GEMM hot loops are pure load->wmma.
// ---------------------------------------------------------------------------

typedef float        v8f   __attribute__((ext_vector_type(8)));
typedef __bf16       v16bf __attribute__((ext_vector_type(16)));
typedef __bf16       v8bf  __attribute__((ext_vector_type(8)));
typedef unsigned int u32x4 __attribute__((ext_vector_type(4)));
typedef int          i32x4 __attribute__((ext_vector_type(4)));
typedef int          i32x8 __attribute__((ext_vector_type(8)));

#define BATCH  2
#define NTOK   4096
#define DMODEL 512
#define NHEAD  8
#define DHEAD  64
#define MTOT   (BATCH * NTOK)          // 8192 rows
// softmax in log2 domain: SCALE * log2(e) = 0.125 * 1.4426950408889634
#define SCALE_LOG2E 0.18033688011112042f

#if __has_builtin(__builtin_amdgcn_tensor_load_to_lds)
#define USE_TDM 1
#else
#define USE_TDM 0
#endif

static __device__ __forceinline__ v8f wmma_bf16(v16bf a, v16bf b, v8f c) {
  return __builtin_amdgcn_wmma_f32_16x16x32_bf16(false, a, false, b, (short)0,
                                                 c, false, false);
}
static __device__ __forceinline__ v16bf cat8(v8bf lo, v8bf hi) {
  v16bf r;
#pragma unroll
  for (int i = 0; i < 8; ++i) { r[i] = lo[i]; r[i + 8] = hi[i]; }
  return r;
}
static __device__ __forceinline__ v8bf ld8(const __bf16* p) {
  return *(const v8bf*)p;
}
static __device__ __forceinline__ v8f ld8f(const float* p) {
  return *(const v8f*)p;
}

#if USE_TDM
// Issue a TDM 2D tile load (bf16 elements) global -> LDS.
// D# layout per CDNA5 ISA.
// tile_d0 = contiguous run (elements), tile_d1 = #rows,
// stride0 = elements between rows, tensor_d0/d1 = bounds (no OOB here).
static __device__ __forceinline__ void tdm_load_2d(
    unsigned lds_byte_off, const __bf16* gptr, unsigned tile_d0,
    unsigned tile_d1, unsigned tensor_d0, unsigned tensor_d1,
    unsigned stride0) {
  const unsigned long long ga = (unsigned long long)gptr;
  u32x4 g0;
  g0[0] = 1u;                                       // count=1, user desc
  g0[1] = lds_byte_off;                             // lds_addr
  g0[2] = (unsigned)(ga & 0xFFFFFFFFull);           // global_addr[31:0]
  g0[3] = (unsigned)((ga >> 32) & 0x1FFFFFFull) | (2u << 30);  // type=2
  i32x8 g1;
  g1[0] = (int)(1u << 16);                          // data_size=1 (2B), mask=0
  g1[1] = (int)((tensor_d0 & 0xFFFFu) << 16);       // tensor_dim0 lo
  g1[2] = (int)(((tensor_d0 >> 16) & 0xFFFFu) | ((tensor_d1 & 0xFFFFu) << 16));
  g1[3] = (int)(((tensor_d1 >> 16) & 0xFFFFu) | (tile_d0 << 16));
  g1[4] = (int)tile_d1;                             // tile_dim1, tile_dim2=0
  g1[5] = (int)stride0;                             // tensor_dim0_stride lo
  g1[6] = 0;
  g1[7] = 0;                                        // tensor_dim1_stride unused
  i32x4 g2; g2[0] = 0; g2[1] = 0; g2[2] = 0; g2[3] = 0;
  i32x4 g3; g3[0] = 0; g3[1] = 0; g3[2] = 0; g3[3] = 0;
#if defined(__clang_major__) && (__clang_major__ >= 23)
  i32x8 g4;
#pragma unroll
  for (int i = 0; i < 8; ++i) g4[i] = 0;
  __builtin_amdgcn_tensor_load_to_lds(g0, g1, g2, g3, g4, 0);
#else
  __builtin_amdgcn_tensor_load_to_lds(g0, g1, g2, g3, 0);
#endif
}
static __device__ __forceinline__ void wait_tensorcnt0() {
#if __has_builtin(__builtin_amdgcn_s_wait_tensorcnt)
  __builtin_amdgcn_s_wait_tensorcnt(0);
#else
  asm volatile("s_wait_tensorcnt 0x0" ::: "memory");
#endif
}
#endif  // USE_TDM

// ---------------------------------------------------------------------------
// fp32 -> bf16 elementwise convert (8 elements per thread). n % 8 == 0.
// ---------------------------------------------------------------------------
__global__ __launch_bounds__(256) void cvt_kernel(const float* __restrict__ in,
                                                  __bf16* __restrict__ out,
                                                  int n) {
  const int i = (blockIdx.x * 256 + threadIdx.x) * 8;
  if (i + 8 <= n) {
    v8f v = ld8f(in + i);
    v8bf o;
#pragma unroll
    for (int j = 0; j < 8; ++j) o[j] = (__bf16)v[j];
    *(v8bf*)(out + i) = o;
  }
}

// ---------------------------------------------------------------------------
// Projection GEMM (all-bf16 operands): Y = Xb[8192,512] @ Wb[512,512]^T
// transposed==0 : Y stored as [B, h, N, d]   (Q, K)
// transposed==1 : Y stored as [B, h, d, N]   (V -> Vt for P*V B-frags)
// Block: 256 thr = 8 waves; wave tile 32(M)x64(N); block tile 256x64.
// ---------------------------------------------------------------------------
__global__ __launch_bounds__(256) void proj_kernel(
    const __bf16* __restrict__ Xb, const __bf16* __restrict__ Wb,
    __bf16* __restrict__ out, int transposed) {
  const int lane = threadIdx.x & 31;
  const int wave = threadIdx.x >> 5;
  const int lm   = lane & 15;
  const int sel  = lane >> 4;
  const int m0   = blockIdx.x * 256 + wave * 32;
  const int n0   = blockIdx.y * 64;

  v8f acc[2][4];
#pragma unroll
  for (int g = 0; g < 2; ++g)
#pragma unroll
    for (int c = 0; c < 4; ++c)
#pragma unroll
      for (int r = 0; r < 8; ++r) acc[g][c][r] = 0.0f;

  const __bf16* arow0 = Xb + (size_t)(m0 + lm) * DMODEL;
  const __bf16* arow1 = Xb + (size_t)(m0 + 16 + lm) * DMODEL;
  for (int k0 = 0; k0 < DMODEL; k0 += 32) {
    v16bf af0 = cat8(ld8(arow0 + k0 + sel * 8), ld8(arow0 + k0 + 16 + sel * 8));
    v16bf af1 = cat8(ld8(arow1 + k0 + sel * 8), ld8(arow1 + k0 + 16 + sel * 8));
#pragma unroll
    for (int c = 0; c < 4; ++c) {
      const __bf16* wrow =
          Wb + (size_t)(n0 + c * 16 + lm) * DMODEL + k0 + sel * 8;
      v16bf bf = cat8(ld8(wrow), ld8(wrow + 16));
      acc[0][c] = wmma_bf16(af0, bf, acc[0][c]);
      acc[1][c] = wmma_bf16(af1, bf, acc[1][c]);
    }
  }

#pragma unroll
  for (int g = 0; g < 2; ++g) {
#pragma unroll
    for (int c = 0; c < 4; ++c) {
#pragma unroll
      for (int r = 0; r < 8; ++r) {
        const int gr = m0 + g * 16 + r + sel * 8;
        const int b  = gr >> 12;
        const int nn = gr & (NTOK - 1);
        const int oc = n0 + c * 16 + lm;
        const int h  = oc >> 6;
        const int dc = oc & (DHEAD - 1);
        const __bf16 v = (__bf16)acc[g][c][r];
        if (transposed)
          out[(((size_t)(b * NHEAD + h) * DHEAD + dc) * NTOK) + nn] = v;
        else
          out[(((size_t)(b * NHEAD + h) * NTOK + nn) * DHEAD) + dc] = v;
      }
    }
  }
}

// ---------------------------------------------------------------------------
// Flash attention. Grid: (B*h = 16, N/64 = 64). Block: 128 thr = 4 waves.
// Wave owns 16 queries x d=64. Streams 64 keys/iteration with online softmax.
// K/Vt tiles TDM-loaded into double-buffered LDS; P bounced through LDS to
// convert C-frag layout -> A-frag layout for the P*V WMMAs.
// ---------------------------------------------------------------------------
__global__ __launch_bounds__(128) void flash_kernel(
    const __bf16* __restrict__ Q, const __bf16* __restrict__ K,
    const __bf16* __restrict__ Vt, __bf16* __restrict__ Obuf) {
  __shared__ __bf16 Ks[2][64 * 64];    // [buf][j][d]   8 KB each
  __shared__ __bf16 Vts[2][64 * 64];   // [buf][d][j]   8 KB each
  __shared__ __bf16 Pl[4][16 * 64];    // per-wave P bounce, 2 KB each

  const int bh   = blockIdx.x;
  const int b    = bh >> 3;
  const int h    = bh & 7;
  const int lane = threadIdx.x & 31;
  const int wave = threadIdx.x >> 5;
  const int lm   = lane & 15;
  const int sel  = lane >> 4;
  const int q0   = blockIdx.y * 64 + wave * 16;

  const __bf16* Qp  = Q  + (size_t)bh * NTOK * DHEAD;
  const __bf16* Kp  = K  + (size_t)bh * NTOK * DHEAD;
  const __bf16* Vtp = Vt + (size_t)bh * DHEAD * NTOK;

  const __bf16* qrow = Qp + (size_t)(q0 + lm) * DHEAD;
  const v16bf qf0 = cat8(ld8(qrow + sel * 8),      ld8(qrow + 16 + sel * 8));
  const v16bf qf1 = cat8(ld8(qrow + 32 + sel * 8), ld8(qrow + 48 + sel * 8));

  float mrun[8], lrun[8];
  v8f o[4];
#pragma unroll
  for (int r = 0; r < 8; ++r) { mrun[r] = -1e30f; lrun[r] = 0.0f; }
#pragma unroll
  for (int c = 0; c < 4; ++c)
#pragma unroll
    for (int r = 0; r < 8; ++r) o[c][r] = 0.0f;

  const int NT = NTOK / 64;

#if USE_TDM
  if (wave == 0) {
    tdm_load_2d((unsigned)(unsigned long long)(const void*)&Ks[0][0],
                Kp, /*tile*/ 64, 64, /*tensor*/ 64, NTOK, /*stride*/ 64);
    tdm_load_2d((unsigned)(unsigned long long)(const void*)&Vts[0][0],
                Vtp, 64, 64, NTOK, 64, NTOK);
  }
#endif

#pragma unroll 1
  for (int t = 0; t < NT; ++t) {
    const int buf = t & 1;
#if USE_TDM
    if (wave == 0) wait_tensorcnt0();
    __syncthreads();                     // tile t visible to all waves
    if ((t + 1) < NT && wave == 0) {     // prefetch tile t+1 into other buffer
      tdm_load_2d((unsigned)(unsigned long long)(const void*)&Ks[buf ^ 1][0],
                  Kp + (size_t)(t + 1) * 64 * DHEAD, 64, 64, 64, NTOK, 64);
      tdm_load_2d((unsigned)(unsigned long long)(const void*)&Vts[buf ^ 1][0],
                  Vtp + (size_t)(t + 1) * 64, 64, 64, NTOK, 64, NTOK);
    }
#else
    __syncthreads();
    {  // cooperative staging fallback: 128 thr x 32 bf16 per array
      const int tt = threadIdx.x;
      const int kr = tt >> 1, kc = (tt & 1) * 32;
      *(v8bf*)(&Ks[buf][kr * 64 + kc])      = ld8(Kp + (size_t)(t * 64 + kr) * DHEAD + kc);
      *(v8bf*)(&Ks[buf][kr * 64 + kc + 8])  = ld8(Kp + (size_t)(t * 64 + kr) * DHEAD + kc + 8);
      *(v8bf*)(&Ks[buf][kr * 64 + kc + 16]) = ld8(Kp + (size_t)(t * 64 + kr) * DHEAD + kc + 16);
      *(v8bf*)(&Ks[buf][kr * 64 + kc + 24]) = ld8(Kp + (size_t)(t * 64 + kr) * DHEAD + kc + 24);
      *(v8bf*)(&Vts[buf][kr * 64 + kc])      = ld8(Vtp + (size_t)kr * NTOK + t * 64 + kc);
      *(v8bf*)(&Vts[buf][kr * 64 + kc + 8])  = ld8(Vtp + (size_t)kr * NTOK + t * 64 + kc + 8);
      *(v8bf*)(&Vts[buf][kr * 64 + kc + 16]) = ld8(Vtp + (size_t)kr * NTOK + t * 64 + kc + 16);
      *(v8bf*)(&Vts[buf][kr * 64 + kc + 24]) = ld8(Vtp + (size_t)kr * NTOK + t * 64 + kc + 24);
    }
    __syncthreads();
#endif

    // ---- S = Q K^T : 16 queries x 64 keys = four 16x16 C-frags ----
    v8f z;
#pragma unroll
    for (int r = 0; r < 8; ++r) z[r] = 0.0f;
    v8f s[4];
#pragma unroll
    for (int i = 0; i < 4; ++i) {
      const __bf16* kr = &Ks[buf][(i * 16 + lm) * 64];
      s[i] = wmma_bf16(qf1, cat8(ld8(kr + 32 + sel * 8), ld8(kr + 48 + sel * 8)), z);
      s[i] = wmma_bf16(qf0, cat8(ld8(kr + sel * 8),      ld8(kr + 16 + sel * 8)),  s[i]);
    }

    // ---- online softmax (log2 domain); rows live across 16 lanes ----
    float alpha[8];
#pragma unroll
    for (int r = 0; r < 8; ++r) {
      float v = -1e30f;
#pragma unroll
      for (int i = 0; i < 4; ++i) {
        s[i][r] *= SCALE_LOG2E;
        v = fmaxf(v, s[i][r]);
      }
#pragma unroll
      for (int off = 1; off < 16; off <<= 1) v = fmaxf(v, __shfl_xor(v, off, 32));
      const float mn = fmaxf(mrun[r], v);
      alpha[r] = exp2f(mrun[r] - mn);
      mrun[r]  = mn;
      float srow = 0.0f;
#pragma unroll
      for (int i = 0; i < 4; ++i) {
        s[i][r] = exp2f(s[i][r] - mn);   // reuse s as P (fp32)
        srow += s[i][r];
      }
#pragma unroll
      for (int off = 1; off < 16; off <<= 1) srow += __shfl_xor(srow, off, 32);
      lrun[r] = lrun[r] * alpha[r] + srow;
    }

    // ---- bounce P through LDS: C-frag layout -> A-frag layout ----
    __bf16* pw = &Pl[wave][0];
#pragma unroll
    for (int i = 0; i < 4; ++i)
#pragma unroll
      for (int r = 0; r < 8; ++r)
        pw[(r + sel * 8) * 64 + i * 16 + lm] = (__bf16)s[i][r];
    const __bf16* prow = pw + lm * 64;
    const v16bf pf0 = cat8(ld8(prow + sel * 8),      ld8(prow + 16 + sel * 8));
    const v16bf pf1 = cat8(ld8(prow + 32 + sel * 8), ld8(prow + 48 + sel * 8));

    // ---- O = O*alpha + P @ V (4 d-chunks x 2 k-halves) ----
#pragma unroll
    for (int c = 0; c < 4; ++c) {
#pragma unroll
      for (int r = 0; r < 8; ++r) o[c][r] *= alpha[r];
      const __bf16* vr = &Vts[buf][(c * 16 + lm) * 64];
      o[c] = wmma_bf16(pf0, cat8(ld8(vr + sel * 8),      ld8(vr + 16 + sel * 8)),  o[c]);
      o[c] = wmma_bf16(pf1, cat8(ld8(vr + 32 + sel * 8), ld8(vr + 48 + sel * 8)), o[c]);
    }
    __syncthreads();                     // done reading buf before overwrite
  }

  // ---- normalize, store as [B, N, h*d] rows for the output GEMM ----
  float inv[8];
#pragma unroll
  for (int r = 0; r < 8; ++r) inv[r] = 1.0f / lrun[r];
#pragma unroll
  for (int c = 0; c < 4; ++c)
#pragma unroll
    for (int r = 0; r < 8; ++r) {
      const int gq = q0 + r + sel * 8;
      Obuf[((size_t)(b * NTOK + gq) * NHEAD + h) * DHEAD + c * 16 + lm] =
          (__bf16)(o[c][r] * inv[r]);
    }
}

// ---------------------------------------------------------------------------
// Output GEMM: out = O[8192,512](bf16) @ Wo(bf16)^T + bo, fp32 result.
// ---------------------------------------------------------------------------
__global__ __launch_bounds__(256) void outproj_kernel(
    const __bf16* __restrict__ A, const __bf16* __restrict__ Wb,
    const float* __restrict__ bias, float* __restrict__ out) {
  const int lane = threadIdx.x & 31;
  const int wave = threadIdx.x >> 5;
  const int lm   = lane & 15;
  const int sel  = lane >> 4;
  const int m0   = blockIdx.x * 256 + wave * 32;
  const int n0   = blockIdx.y * 64;

  v8f acc[2][4];
#pragma unroll
  for (int g = 0; g < 2; ++g)
#pragma unroll
    for (int c = 0; c < 4; ++c)
#pragma unroll
      for (int r = 0; r < 8; ++r) acc[g][c][r] = 0.0f;

  const __bf16* arow0 = A + (size_t)(m0 + lm) * DMODEL;
  const __bf16* arow1 = A + (size_t)(m0 + 16 + lm) * DMODEL;
  for (int k0 = 0; k0 < DMODEL; k0 += 32) {
    v16bf af0 = cat8(ld8(arow0 + k0 + sel * 8), ld8(arow0 + k0 + 16 + sel * 8));
    v16bf af1 = cat8(ld8(arow1 + k0 + sel * 8), ld8(arow1 + k0 + 16 + sel * 8));
#pragma unroll
    for (int c = 0; c < 4; ++c) {
      const __bf16* wrow =
          Wb + (size_t)(n0 + c * 16 + lm) * DMODEL + k0 + sel * 8;
      v16bf bf = cat8(ld8(wrow), ld8(wrow + 16));
      acc[0][c] = wmma_bf16(af0, bf, acc[0][c]);
      acc[1][c] = wmma_bf16(af1, bf, acc[1][c]);
    }
  }

#pragma unroll
  for (int g = 0; g < 2; ++g)
#pragma unroll
    for (int c = 0; c < 4; ++c) {
      const float bv = bias[n0 + c * 16 + lm];
#pragma unroll
      for (int r = 0; r < 8; ++r) {
        const int gr = m0 + g * 16 + r + sel * 8;
        out[(size_t)gr * DMODEL + n0 + c * 16 + lm] = acc[g][c][r] + bv;
      }
    }
}

// ---------------------------------------------------------------------------
extern "C" void kernel_launch(void* const* d_in, const int* in_sizes, int n_in,
                              void* d_out, int out_size, void* d_ws,
                              size_t ws_size, hipStream_t stream) {
  const float* x  = (const float*)d_in[0];
  const float* Wq = (const float*)d_in[1];
  const float* Wk = (const float*)d_in[2];
  const float* Wv = (const float*)d_in[3];
  const float* Wo = (const float*)d_in[4];
  const float* bo = (const float*)d_in[5];
  float* out = (float*)d_out;

  const size_t XN = (size_t)MTOT * DMODEL;     // 4,194,304
  const size_t WN = (size_t)DMODEL * DMODEL;   // 262,144
  __bf16* xb  = (__bf16*)d_ws;                 // [8192, 512]
  __bf16* Wqb = xb + XN;
  __bf16* Wkb = Wqb + WN;
  __bf16* Wvb = Wkb + WN;
  __bf16* Wob = Wvb + WN;
  __bf16* Qb  = Wob + WN;                      // [B,h,N,d]
  __bf16* Kb  = Qb + XN;                       // [B,h,N,d]
  __bf16* Vtb = Kb + XN;                       // [B,h,d,N]
  __bf16* Ob  = Vtb + XN;                      // [B,N,h*d]
  // total ws use: (5*XN + 4*WN) * 2 bytes = 44 MB

  cvt_kernel<<<dim3((int)(XN / 8 / 256)), 256, 0, stream>>>(x, xb, (int)XN);
  cvt_kernel<<<dim3((int)(WN / 8 / 256)), 256, 0, stream>>>(Wq, Wqb, (int)WN);
  cvt_kernel<<<dim3((int)(WN / 8 / 256)), 256, 0, stream>>>(Wk, Wkb, (int)WN);
  cvt_kernel<<<dim3((int)(WN / 8 / 256)), 256, 0, stream>>>(Wv, Wvb, (int)WN);
  cvt_kernel<<<dim3((int)(WN / 8 / 256)), 256, 0, stream>>>(Wo, Wob, (int)WN);

  const dim3 gGemm(MTOT / 256, DMODEL / 64);
  proj_kernel<<<gGemm, 256, 0, stream>>>(xb, Wqb, Qb, 0);
  proj_kernel<<<gGemm, 256, 0, stream>>>(xb, Wkb, Kb, 0);
  proj_kernel<<<gGemm, 256, 0, stream>>>(xb, Wvb, Vtb, 1);

  const dim3 gAttn(BATCH * NHEAD, NTOK / 64);
  flash_kernel<<<gAttn, 128, 0, stream>>>(Qb, Kb, Vtb, Ob);

  outproj_kernel<<<gGemm, 256, 0, stream>>>(Ob, Wob, bo, out);
}
